// SAU_81346680586754
// MI455X (gfx1250) — matrix operand order repaired and checked by reading, not compiled, not run on hardware
//
#include <hip/hip_runtime.h>
#include <hip/hip_bf16.h>

// CARAFE-style content-aware upsampling for gfx1250 (MI455X).
// Conv stages: f16 WMMA GEMMs (v_wmma_f32_16x16x32_f16), f32 accumulate.
// Activation slabs staged into LDS via the Tensor Data Mover (tensor_load_to_lds,
// TENSORcnt). N=8, C=128, H=W=64, S=2, K=5, NK=100.

typedef __attribute__((ext_vector_type(16))) _Float16 v16h;
typedef __attribute__((ext_vector_type(8)))  _Float16 v8h;
typedef __attribute__((ext_vector_type(8)))  float    v8f;
typedef __attribute__((ext_vector_type(4)))  unsigned int u32x4;
typedef __attribute__((ext_vector_type(8)))  int i32x8;
typedef __attribute__((ext_vector_type(4)))  int i32x4;

__device__ __forceinline__ v8f wmma_f16(v16h a, v16h b, v8f c) {
  return __builtin_amdgcn_wmma_f32_16x16x32_f16(
      false, a, false, b, (short)0, c, false, false);
}

// 1-D TDM copy: n8 elements of 8 bytes, global -> LDS.
// D# group0: count=1 | lds_addr | global_addr(57b) | type=2
// D# group1: data_size=3 (8B); tensor_dim0 = tile_dim0 = n8; tensor_dim1 = 1
__device__ __forceinline__ void tdm_load_1d(unsigned lds_off, const void* gptr,
                                            unsigned n8) {
  unsigned long long ga = (unsigned long long)gptr;
  u32x4 g0;
  g0[0] = 1u;                                            // count=1, user mode
  g0[1] = lds_off;                                       // LDS byte address
  g0[2] = (unsigned)ga;                                  // global_addr[31:0]
  g0[3] = (unsigned)((ga >> 32) & 0x1FFFFFFu) | (2u << 30);  // [56:32] | type=2
  i32x8 g1;
  g1[0] = (int)(3u << 16);                               // data_size = 8B
  g1[1] = (int)((n8 & 0xFFFFu) << 16);                   // tensor_dim0[15:0]
  g1[2] = (int)((n8 >> 16) | (1u << 16));                // dim0[31:16] | dim1=1
  g1[3] = (int)((n8 & 0xFFFFu) << 16);                   // tile_dim0 = n8
  g1[4] = 0;                                             // tile_dim1/2 unused
  g1[5] = (int)n8;                                       // tensor_dim0_stride
  g1[6] = 0;
  g1[7] = 0;
  i32x4 gz4 = {0, 0, 0, 0};
  i32x8 gz8 = {0, 0, 0, 0, 0, 0, 0, 0};
  // clang-23 / therock toolchain: 6-argument form
  __builtin_amdgcn_tensor_load_to_lds(g0, g1, gz4, gz4, gz8, 0);
}

// ---------------------------------------------------------------- prep kernels

__global__ __launch_bounds__(256) void k_zero4(uint4* __restrict__ p, long n4) {
  long i = (long)blockIdx.x * 256 + threadIdx.x;
  long stride = (long)gridDim.x * 256;
  uint4 z = {0u, 0u, 0u, 0u};
  for (; i < n4; i += stride) p[i] = z;
}

// x (8,128,64,64) f32 NCHW -> xh f16 NHWC zero-padded [8][66][66][128]
__global__ __launch_bounds__(256) void k_pack_x(const float* __restrict__ x,
                                                _Float16* __restrict__ xh) {
  int tid = blockIdx.x * 256 + threadIdx.x;        // 4,194,304 total
  int c = tid & 127, w = (tid >> 7) & 63, h = (tid >> 13) & 63, n = tid >> 19;
  float v = x[(((n << 7) + c) << 12) + (h << 6) + w];
  xh[((n * 66 + h + 1) * 66 + (w + 1)) * 128 + c] = (_Float16)v;
}

// fragment K-index map: contiguous order k = 16*(lane/16) + e (matches B loads)
__device__ __forceinline__ int kmapA(int lane, int e) {
  return e + ((lane >> 4) << 4);
}

// w_enc (64,128,3,3) -> fragment-major f16, K = tap*128 + ic, 36 ktiles x 4 octiles
__global__ __launch_bounds__(256) void k_pack_wenc(const float* __restrict__ w,
                                                   _Float16* __restrict__ wf) {
  int tid = blockIdx.x * 256 + threadIdx.x;        // 36*4*512 = 73728
  int e = tid & 15, lane = (tid >> 4) & 31, f = tid >> 9;
  int kt = f >> 2, ot = f & 3;
  int oc = (ot << 4) + (lane & 15);
  int k = (kt << 5) + kmapA(lane, e);
  int tap = k >> 7, ic = k & 127;
  int dy = tap / 3, dx = tap - dy * 3;
  wf[tid] = (_Float16)w[((oc * 128 + ic) * 3 + dy) * 3 + dx];
}

// w_g1 (64,64,3,3) -> fragment-major, K = tap*64 + ic, 18 ktiles x 4 octiles
__global__ __launch_bounds__(256) void k_pack_wg1(const float* __restrict__ w,
                                                  _Float16* __restrict__ wf) {
  int tid = blockIdx.x * 256 + threadIdx.x;        // 18*4*512 = 36864
  int e = tid & 15, lane = (tid >> 4) & 31, f = tid >> 9;
  int kt = f >> 2, ot = f & 3;
  int oc = (ot << 4) + (lane & 15);
  int k = (kt << 5) + kmapA(lane, e);
  int tap = k >> 6, ic = k & 63;
  int dy = tap / 3, dx = tap - dy * 3;
  wf[tid] = (_Float16)w[((oc * 64 + ic) * 3 + dy) * 3 + dx];
}

// 1x1 weights (100,64) -> fragment-major, 2 ktiles x 7 octiles (pad oc>=100 with 0)
__global__ __launch_bounds__(256) void k_pack_w1x1(const float* __restrict__ w,
                                                   _Float16* __restrict__ wf) {
  int tid = blockIdx.x * 256 + threadIdx.x;        // 2*7*512 = 7168
  int e = tid & 15, lane = (tid >> 4) & 31, f = tid >> 9;
  int kt = f / 7, ot = f - kt * 7;                 // f = kt*7 + ot
  int oc = (ot << 4) + (lane & 15);
  int ic = (kt << 5) + kmapA(lane, e);
  float v = (oc < 100) ? w[oc * 64 + ic] : 0.f;
  wf[tid] = (_Float16)v;
}

// fold BN + encoder bias into per-channel scale/shift
__global__ void k_bn(const float* __restrict__ be, const float* __restrict__ g,
                     const float* __restrict__ bt, const float* __restrict__ mean,
                     const float* __restrict__ var, float* __restrict__ scale,
                     float* __restrict__ shift) {
  int i = threadIdx.x;                             // 64 threads
  float sc = g[i] * rsqrtf(var[i] + 1e-5f);
  scale[i] = sc;
  shift[i] = (be[i] - mean[i]) * sc + bt[i];
}

// ------------------------------------------------------------- WMMA conv GEMMs
// Block = 128 threads (4 waves) = one output row of 64 positions.
// TDM stages the 3 padded input rows into LDS; B-fragments come from LDS,
// A-fragments are double-buffered in registers so loads overlap WMMAs.

// content = ReLU(BN(conv3x3(x)))  -> chh f16 NHWC zero-padded [8][66][66][64]
__global__ __launch_bounds__(128) void k_encoder(const _Float16* __restrict__ xh,
                                                 const _Float16* __restrict__ wf,
                                                 const float* __restrict__ scale,
                                                 const float* __restrict__ shift,
                                                 _Float16* __restrict__ chh) {
  __shared__ __align__(32) _Float16 sx[3 * 66 * 128];    // 50,688 B
  const int lane = threadIdx.x & 31;
  const int wave = threadIdx.x >> 5;                     // 0..3
  const int n = blockIdx.x >> 6;
  const int h = blockIdx.x & 63;
  const int w0 = wave << 4;
  const int p = lane & 15, hi = lane >> 4;

  if (threadIdx.x < 32) {
    const _Float16* src = xh + (size_t)((n * 66 + h) * 66) * 128;
    tdm_load_1d((unsigned)(uintptr_t)sx, src, (3 * 66 * 128 * 2) / 8);
    __builtin_amdgcn_s_wait_tensorcnt(0);
  }
  __syncthreads();

  v8f acc[4] = {{}, {}, {}, {}};
  const _Float16* afb = wf + lane * 16;
  __builtin_prefetch(afb, 0, 1);
  v16h aF[2][4];
#pragma unroll
  for (int ot = 0; ot < 4; ++ot) aF[0][ot] = *(const v16h*)(afb + ot * 512);
#pragma unroll
  for (int i = 0; i < 36; ++i) {                         // chunk = (tap, cc)
    const int cur = i & 1;
    if (i + 1 < 36) {
#pragma unroll
      for (int ot = 0; ot < 4; ++ot)
        aF[cur ^ 1][ot] = *(const v16h*)(afb + ((i + 1) * 4 + ot) * 512);
    }
    const int tap = i >> 2, cc = i & 3;
    const int dy = tap / 3, dx = tap - dy * 3;
    const v16h b =
        *(const v16h*)(&sx[(dy * 66 + (w0 + p + dx)) * 128 + cc * 32 + hi * 16]);
#pragma unroll
    for (int ot = 0; ot < 4; ++ot) acc[ot] = wmma_f16(aF[cur][ot], b, acc[ot]);
  }

  _Float16* dst = chh + ((n * 66 + h + 1) * 66 + (w0 + p + 1)) * 64;
#pragma unroll
  for (int ot = 0; ot < 4; ++ot) {
    int oc0 = ot * 16 + hi * 8;
    v8h o;
#pragma unroll
    for (int r = 0; r < 8; ++r) {
      float v = acc[ot][r] * scale[oc0 + r] + shift[oc0 + r];
      o[r] = (_Float16)(v > 0.f ? v : 0.f);
    }
    *(v8h*)(dst + oc0) = o;
  }
}

// g1 = ReLU(conv3x3(content))  -> g1h f16 NHWC [8][64][64][64]
__global__ __launch_bounds__(128) void k_gate1(const _Float16* __restrict__ chh,
                                               const _Float16* __restrict__ wf,
                                               const float* __restrict__ bg1,
                                               _Float16* __restrict__ g1h) {
  __shared__ __align__(32) _Float16 sc[3 * 66 * 64];     // 25,344 B
  const int lane = threadIdx.x & 31;
  const int wave = threadIdx.x >> 5;
  const int n = blockIdx.x >> 6;
  const int h = blockIdx.x & 63;
  const int w0 = wave << 4;
  const int p = lane & 15, hi = lane >> 4;

  if (threadIdx.x < 32) {
    const _Float16* src = chh + (size_t)((n * 66 + h) * 66) * 64;
    tdm_load_1d((unsigned)(uintptr_t)sc, src, (3 * 66 * 64 * 2) / 8);
    __builtin_amdgcn_s_wait_tensorcnt(0);
  }
  __syncthreads();

  v8f acc[4] = {{}, {}, {}, {}};
  const _Float16* afb = wf + lane * 16;
  v16h aF[2][4];
#pragma unroll
  for (int ot = 0; ot < 4; ++ot) aF[0][ot] = *(const v16h*)(afb + ot * 512);
#pragma unroll
  for (int i = 0; i < 18; ++i) {                         // chunk = (tap, cc)
    const int cur = i & 1;
    if (i + 1 < 18) {
#pragma unroll
      for (int ot = 0; ot < 4; ++ot)
        aF[cur ^ 1][ot] = *(const v16h*)(afb + ((i + 1) * 4 + ot) * 512);
    }
    const int tap = i >> 1, cc = i & 1;
    const int dy = tap / 3, dx = tap - dy * 3;
    const v16h b =
        *(const v16h*)(&sc[(dy * 66 + (w0 + p + dx)) * 64 + cc * 32 + hi * 16]);
#pragma unroll
    for (int ot = 0; ot < 4; ++ot) acc[ot] = wmma_f16(aF[cur][ot], b, acc[ot]);
  }

  _Float16* dst = g1h + ((n << 12) + h * 64 + (w0 + p)) * 64;
#pragma unroll
  for (int ot = 0; ot < 4; ++ot) {
    int oc0 = ot * 16 + hi * 8;
    v8h o;
#pragma unroll
    for (int r = 0; r < 8; ++r) {
      float v = acc[ot][r] + bg1[oc0 + r];
      o[r] = (_Float16)(v > 0.f ? v : 0.f);
    }
    *(v8h*)(dst + oc0) = o;
  }
}

// kernel-pred 1x1, gate 1x1 + sigmoid, product, softmax over 100 chans
// -> ker f32 (8,100,64,64)
__global__ __launch_bounds__(256) void k_predict(
    const _Float16* __restrict__ chh, const _Float16* __restrict__ g1h,
    const _Float16* __restrict__ wkp, const _Float16* __restrict__ wg2,
    const float* __restrict__ bkp, const float* __restrict__ bg2,
    float* __restrict__ ker) {
  const int lane = threadIdx.x & 31;
  const int tile = blockIdx.x * 8 + (threadIdx.x >> 5);  // 16384 pos-tiles
  const int n = tile >> 8;
  const int h = (tile >> 2) & 63;
  const int w0 = (tile & 3) << 4;
  const int p = lane & 15, hi = lane >> 4;

  const _Float16* cb =
      chh + ((n * 66 + h + 1) * 66 + (w0 + p + 1)) * 64 + hi * 16;
  v16h bc0 = *(const v16h*)cb;
  v16h bc1 = *(const v16h*)(cb + 32);
  const _Float16* gb = g1h + ((n << 12) + h * 64 + (w0 + p)) * 64 + hi * 16;
  v16h bg0 = *(const v16h*)gb;
  v16h bg1 = *(const v16h*)(gb + 32);

  float s[56];
  float m = -1e30f;
#pragma unroll
  for (int ot = 0; ot < 7; ++ot) {
    v8f aK = {};
    v8f aG = {};
    const _Float16* akf = wkp + ot * 512 + lane * 16;
    const _Float16* agf = wg2 + ot * 512 + lane * 16;
    v16h a;
    a = *(const v16h*)(akf);            aK = wmma_f16(a, bc0, aK);
    a = *(const v16h*)(akf + 7 * 512);  aK = wmma_f16(a, bc1, aK);
    a = *(const v16h*)(agf);            aG = wmma_f16(a, bg0, aG);
    a = *(const v16h*)(agf + 7 * 512);  aG = wmma_f16(a, bg1, aG);
#pragma unroll
    for (int r = 0; r < 8; ++r) {
      int chan = ot * 16 + hi * 8 + r;
      float v;
      if (chan < 100) {
        float g = 1.f / (1.f + __expf(-(aG[r] + bg2[chan])));
        v = (aK[r] + bkp[chan]) * g;
      } else {
        v = -1e30f;  // padded channels drop out of softmax
      }
      s[ot * 8 + r] = v;
      m = fmaxf(m, v);
    }
  }
  // rows of one position live in lanes p and p+16 (C/D layout) -> xor-16 reduce
  m = fmaxf(m, __shfl_xor(m, 16, 32));
  float sum = 0.f;
#pragma unroll
  for (int i = 0; i < 56; ++i) {
    float e = __expf(s[i] - m);
    s[i] = e;
    sum += e;
  }
  sum += __shfl_xor(sum, 16, 32);
  float inv = 1.f / sum;
  float* kb = ker + (size_t)n * 100 * 4096 + h * 64 + (w0 + p);
#pragma unroll
  for (int ot = 0; ot < 7; ++ot) {
#pragma unroll
    for (int r = 0; r < 8; ++r) {
      int chan = ot * 16 + hi * 8 + r;
      if (chan < 100) kb[(size_t)chan * 4096] = s[ot * 8 + r] * inv;
    }
  }
}

// ------------------------------------------------------------------ reassembly

// out[n,c,2h+s,2w+t] = sum_{i,j} x[n,c,clamp(h+i-2),clamp(w+j-2)]
//                                * ker[n,(i*5+j)*4+s*2+t,h,w]
__global__ __launch_bounds__(256) void k_reassemble(const float* __restrict__ x,
                                                    const float* __restrict__ ker,
                                                    float* __restrict__ out) {
  __shared__ float kl[64 * 104];
  const int n = blockIdx.y;
  const int h = blockIdx.x;
  const int tid = threadIdx.x;
  for (int e = tid; e < 6400; e += 256) {
    int chan = e >> 6, w = e & 63;
    kl[w * 104 + chan] = ker[((size_t)(n * 100 + chan) * 64 + h) * 64 + w];
  }
  __syncthreads();
  const int w = tid & 63;
  const int c0 = tid >> 6;
#pragma unroll 1
  for (int c = c0; c < 128; c += 4) {
    const float* xb = x + (size_t)(n * 128 + c) * 4096;
    float o0 = 0.f, o1 = 0.f, o2 = 0.f, o3 = 0.f;
#pragma unroll
    for (int i = 0; i < 5; ++i) {
      int hr = h - 2 + i;
      hr = hr < 0 ? 0 : (hr > 63 ? 63 : hr);
      const float* xr = xb + hr * 64;
#pragma unroll
      for (int j = 0; j < 5; ++j) {
        int wr = w - 2 + j;
        wr = wr < 0 ? 0 : (wr > 63 ? 63 : wr);
        float xv = xr[wr];
        const float4 kv = *(const float4*)&kl[w * 104 + (i * 5 + j) * 4];
        o0 += xv * kv.x;
        o1 += xv * kv.y;
        o2 += xv * kv.z;
        o3 += xv * kv.w;
      }
    }
    float* ob = out + ((size_t)(n * 128 + c) * 128 + 2 * h) * 128 + 2 * w;
    *(float2*)(ob) = make_float2(o0, o1);
    *(float2*)(ob + 128) = make_float2(o2, o3);
  }
}

// ---------------------------------------------------------------------- launch

extern "C" void kernel_launch(void* const* d_in, const int* in_sizes, int n_in,
                              void* d_out, int out_size, void* d_ws, size_t ws_size,
                              hipStream_t stream) {
  (void)in_sizes; (void)n_in; (void)out_size; (void)ws_size;
  const float* x     = (const float*)d_in[0];
  const float* w_enc = (const float*)d_in[1];
  const float* b_enc = (const float*)d_in[2];
  const float* gamma = (const float*)d_in[3];
  const float* beta  = (const float*)d_in[4];
  const float* mean  = (const float*)d_in[5];
  const float* var   = (const float*)d_in[6];
  const float* w_kp  = (const float*)d_in[7];
  const float* b_kp  = (const float*)d_in[8];
  const float* w_g1  = (const float*)d_in[9];
  const float* b_g1  = (const float*)d_in[10];
  const float* w_g2  = (const float*)d_in[11];
  const float* b_g2  = (const float*)d_in[12];
  float* out = (float*)d_out;

  char* ws = (char*)d_ws;
  size_t off = 0;
  auto take = [&](size_t b) {
    char* r = ws + off;
    off = (off + b + 255) & ~(size_t)255;
    return r;
  };
  const size_t xh_bytes  = 8ull * 66 * 66 * 128 * 2;  // padded NHWC f16 x
  const size_t chh_bytes = 8ull * 66 * 66 * 64 * 2;   // padded NHWC f16 content
  _Float16* xh    = (_Float16*)take(xh_bytes);
  _Float16* chh   = (_Float16*)take(chh_bytes);       // contiguous with xh
  _Float16* g1h   = (_Float16*)take(8ull * 64 * 64 * 64 * 2);
  float*    kerb  = (float*)take(8ull * 100 * 64 * 64 * 4);
  _Float16* wencF = (_Float16*)take(36ull * 4 * 512 * 2);
  _Float16* wg1F  = (_Float16*)take(18ull * 4 * 512 * 2);
  _Float16* wkpF  = (_Float16*)take(2ull * 7 * 512 * 2);
  _Float16* wg2F  = (_Float16*)take(2ull * 7 * 512 * 2);
  float*    bnsc  = (float*)take(64 * 4);
  float*    bnsh  = (float*)take(64 * 4);

  // zero the two zero-padded f16 buffers (contiguous region)
  long n4 = (long)((xh_bytes + chh_bytes) / 16);
  k_zero4<<<3267, 256, 0, stream>>>((uint4*)xh, n4);

  k_pack_x<<<16384, 256, 0, stream>>>(x, xh);
  k_pack_wenc<<<288, 256, 0, stream>>>(w_enc, wencF);
  k_pack_wg1<<<144, 256, 0, stream>>>(w_g1, wg1F);
  k_pack_w1x1<<<28, 256, 0, stream>>>(w_kp, wkpF);
  k_pack_w1x1<<<28, 256, 0, stream>>>(w_g2, wg2F);
  k_bn<<<1, 64, 0, stream>>>(b_enc, gamma, beta, mean, var, bnsc, bnsh);

  k_encoder<<<512, 128, 0, stream>>>(xh, wencF, bnsc, bnsh, chh);
  k_gate1<<<512, 128, 0, stream>>>(chh, wg1F, b_g1, g1h);
  k_predict<<<2048, 256, 0, stream>>>(chh, g1h, wkpF, wg2F, b_kp, b_g2, kerb);
  k_reassemble<<<dim3(64, 8), 256, 0, stream>>>(x, kerb, out);
}
